// SpaFGAN_1468878815285
// MI455X (gfx1250) — matrix operand order repaired and checked by reading, not compiled, not run on hardware
//
#include <hip/hip_runtime.h>
#include <hip/hip_fp16.h>

// CDNA5 / gfx1250 WMMA fragment types (wave32)
typedef __attribute__((ext_vector_type(16))) _Float16 v16h;
typedef __attribute__((ext_vector_type(8)))  float    v8f;

#define LN_EPS 1e-5f
#define SLOPE  0.2f
#define CDIV(a, b) (((a) + (b) - 1) / (b))

// gfx1250 async global->LDS path (guarded: falls back to register staging)
#if defined(__AMDGCN__) && __has_builtin(__builtin_amdgcn_global_load_async_to_lds_b128) && \
    __has_builtin(__builtin_amdgcn_s_wait_asynccnt)
#define USE_ASYNC_LDS 1
#else
#define USE_ASYNC_LDS 0
#endif

#if USE_ASYNC_LDS
// Builtin signature (from hipcc diagnostic): params are pointers to
// 16-byte GCC-style int vectors in AS1 (global) / AS3 (LDS).
typedef int v4i_raw __attribute__((vector_size(16)));
// integer round-trip casts: int->ptr is always legal; generic LDS pointer's
// low 32 bits are the LDS offset (AS3 pointers are 32-bit).
#define AS1_V4I(p) ((__attribute__((address_space(1))) v4i_raw*)(unsigned long long)(p))
#define AS3_V4I(p) ((__attribute__((address_space(3))) v4i_raw*)(unsigned)(unsigned long long)(p))
#endif

// ---- ordered-uint encoding of float so unsigned atomicMax == float max ----
__device__ __forceinline__ unsigned fenc(float f) {
  unsigned u = __float_as_uint(f);
  return (u & 0x80000000u) ? ~u : (u | 0x80000000u);
}
__device__ __forceinline__ float fdec(unsigned e) {
  return (e & 0x80000000u) ? __uint_as_float(e & 0x7fffffffu)
                           : __uint_as_float(~e);
}
#define ENC_NEGINF 0x007FFFFFu  // fenc(-inf)

__device__ __forceinline__ float leaky(float v) { return v > 0.f ? v : SLOPE * v; }
__device__ __forceinline__ float elu(float v)   { return v > 0.f ? v : (expf(v) - 1.f); }

// ---------------------------------------------------------------------------
// fill kernel (init max buffers / zero accumulators)
// ---------------------------------------------------------------------------
__global__ void fill_u32(unsigned* __restrict__ p, unsigned v, long long cnt) {
  long long i = (long long)blockIdx.x * blockDim.x + threadIdx.x;
  if (i < cnt) p[i] = v;
}

// ---------------------------------------------------------------------------
// H[nrows x C] = X[nrows x 64] @ W[64 x C]  via v_wmma_f32_16x16x32_f16
//
// Block = 256 threads = 8 waves. Each block owns 128 rows of X (async-staged
// into LDS as raw f32 via global_load_async_to_lds_b128) and one 16-column
// slice of W (staged into LDS as f16, layout [col][k] so a lane's B fragment
// is one contiguous 32B LDS read). Each wave computes one 16x16 output tile;
// K=64 => 2 WMMA issues.
//
// Fragment layouts (ISA 7.12.2, wave32):
//   A (f16 16x32): lane L: row = L%16, g = L/16, elem e: K = e + 16g (this code's order)
//   B (f16 32x16): lane L: col = L%16, g = L/16, elem e: K = e + 16g (same order as A, not ISA table order)
//   C/D (f32 16x16): lane L: col = L%16, vgpr r: row = r + 8*(L/16)
// ---------------------------------------------------------------------------
template <int C>
__global__ void gemm_wmma(const float* __restrict__ X, const float* __restrict__ W,
                          float* __restrict__ H, int nrows) {
  __shared__ alignas(32) float    Xl[128 * 64];  // 32 KB raw f32 X tile
  __shared__ alignas(32) _Float16 Wl[16 * 64];   // 2 KB f16 W tile, [col][k]

  const int t        = threadIdx.x;
  const int lane     = t & 31;
  const int wave     = t >> 5;
  const int blockRow = blockIdx.x * 128;
  const int colBase  = blockIdx.y * 16;

  // ---- stage X[blockRow..blockRow+127][0..64) into LDS ----
  // 8192 floats = 256 threads x 8 x float4 (16B granules)
#pragma unroll
  for (int i = 0; i < 8; ++i) {
    int id4 = (i * 256 + t) * 4;                 // float index, 16B aligned
    int r   = id4 >> 6;                          // row within tile
    int gr  = min(blockRow + r, nrows - 1);      // clamp ragged tail
    const float* gp = X + (long long)gr * 64 + (id4 & 63);
#if USE_ASYNC_LDS
    __builtin_amdgcn_global_load_async_to_lds_b128(AS1_V4I(gp), AS3_V4I(Xl + id4), 0, 0);
#else
    *(float4*)(Xl + id4) = *(const float4*)gp;
#endif
  }

  // ---- stage W[:, colBase..colBase+16) as f16 into LDS, layout [c][k] ----
  {
    int id = t * 4;                              // 1024 halves total
    int c = id >> 6, k = id & 63;
#pragma unroll
    for (int j = 0; j < 4; ++j)
      Wl[id + j] = (_Float16)W[(k + j) * C + colBase + c];
  }

#if USE_ASYNC_LDS
  __builtin_amdgcn_s_wait_asynccnt(0);
#endif
  __syncthreads();

  const int rowTile = blockRow + wave * 16;
  if (rowTile < nrows) {                         // wave-uniform: EXEC stays all-1s
    const int m = lane & 15;
    const int g = lane >> 4;
    const float* xr0 = Xl + (wave * 16 + m) * 64;

    v8f acc = {};
#pragma unroll
    for (int ks = 0; ks < 2; ++ks) {
      v16h a;
      const float* xr = xr0 + ks * 32 + g * 16;
#pragma unroll
      for (int e = 0; e < 8; ++e) {
        a[e]     = (_Float16)xr[e];              // K = 16g + e     (+32*ks)
        a[e + 8] = (_Float16)xr[e + 8];          // K = 16g + 8 + e (+32*ks)
      }
      v16h b = *(const v16h*)(Wl + m * 64 + ks * 32 + g * 16);  // K = 16g + e (+32*ks)

      acc = __builtin_amdgcn_wmma_f32_16x16x32_f16(
          /*neg_a=*/false, a, /*neg_b=*/false, b,
          /*c_mod=*/(short)0, acc, /*reuse_a=*/false, /*reuse_b=*/false);
    }

    if (rowTile + 16 <= nrows) {                 // full tile: straight-line stores
#pragma unroll
      for (int r = 0; r < 8; ++r)
        H[(long long)(rowTile + r + 8 * g) * C + colBase + m] = acc[r];
    } else {                                     // ragged tail (unused when N%16==0)
#pragma unroll
      for (int r = 0; r < 8; ++r) {
        int rr = rowTile + r + 8 * g;
        if (rr < nrows) H[(long long)rr * C + colBase + m] = acc[r];
      }
    }
  }
}

// ---------------------------------------------------------------------------
// per-node attention coefficients: als[n,h] = <h[n,h,:], a_src[h,:]>, same dst
// ---------------------------------------------------------------------------
__global__ void node_alpha(const float* __restrict__ h,
                           const float* __restrict__ a_src,
                           const float* __restrict__ a_dst,
                           float* __restrict__ als, float* __restrict__ ald,
                           int Nn, int Hh) {
  int gid = blockIdx.x * blockDim.x + threadIdx.x;
  if (gid >= Nn * Hh) return;
  int n = gid / Hh, hh = gid - n * Hh;
  const float* hp = h + ((long long)n * Hh + hh) * 32;
  const float* as = a_src + hh * 32;
  const float* ad = a_dst + hh * 32;
  float s = 0.f, d = 0.f;
#pragma unroll
  for (int c = 0; c < 32; ++c) { s += hp[c] * as[c]; d += hp[c] * ad[c]; }
  als[gid] = s;
  ald[gid] = d;
}

// ---------------------------------------------------------------------------
// pass 1: segment max of leaky-relu logits over destinations (incl. self loops)
// ---------------------------------------------------------------------------
__global__ void edge_max(const long long* __restrict__ ei, int E, int Nn, int Hh,
                         const float* __restrict__ als, const float* __restrict__ ald,
                         unsigned* __restrict__ mx) {
  long long gid = (long long)blockIdx.x * blockDim.x + threadIdx.x;
  long long tot = (long long)(E + Nn) * Hh;
  if (gid >= tot) return;
  int h = (int)(gid % Hh);
  long long i = gid / Hh;
  int s, d;
  if (i < E) {
    __builtin_prefetch(ei + i + 2048, 0, 0);     // global_prefetch_b8 on edge stream
    s = (int)ei[i];
    d = (int)ei[E + i];
  } else {
    s = d = (int)(i - E);
  }
  float l = leaky(als[s * Hh + h] + ald[d * Hh + h]);
  atomicMax(mx + d * Hh + h, fenc(l));
}

// ---------------------------------------------------------------------------
// pass 2: e = exp(logit - max[dst]); denom[dst] += e; agg[dst,:] += h[src,:]*e
// 16 (or 8) threads per edge: each handles 4 channels; part 0 of each head
// also accumulates the denominator. All atomics land in L2 (buffers ~25 MB).
// ---------------------------------------------------------------------------
__global__ void edge_acc(const long long* __restrict__ ei, int E, int Nn, int Hh,
                         const float* __restrict__ als, const float* __restrict__ ald,
                         const unsigned* __restrict__ mx,
                         const float* __restrict__ hfeat,
                         float* __restrict__ agg, float* __restrict__ denom) {
  const int parts = Hh * 8;                      // chunks of 4 channels
  long long gid = (long long)blockIdx.x * blockDim.x + threadIdx.x;
  long long tot = (long long)(E + Nn) * parts;
  if (gid >= tot) return;
  int p = (int)(gid % parts);
  long long i = gid / parts;
  int s, d;
  if (i < E) {
    s = (int)ei[i];
    d = (int)ei[E + i];
  } else {
    s = d = (int)(i - E);
  }
  int h = p >> 3;
  float l = leaky(als[s * Hh + h] + ald[d * Hh + h]);
  float w = expf(l - fdec(mx[d * Hh + h]));
  if ((p & 7) == 0) atomicAdd(denom + d * Hh + h, w);
  int cb = p * 4;
  const float* hs = hfeat + (long long)s * (Hh * 32) + cb;
  float* ad = agg + (long long)d * (Hh * 32) + cb;
  atomicAdd(ad + 0, hs[0] * w);
  atomicAdd(ad + 1, hs[1] * w);
  atomicAdd(ad + 2, hs[2] * w);
  atomicAdd(ad + 3, hs[3] * w);
}

// ---------------------------------------------------------------------------
// layer-1 epilogue: normalize softmax, +bias, LayerNorm(64), ELU -> x2
// ---------------------------------------------------------------------------
__global__ void finalize1(const float* __restrict__ agg, const float* __restrict__ denom,
                          const float* __restrict__ b, const float* __restrict__ g,
                          const float* __restrict__ be, float* __restrict__ x2, int Nn) {
  int n = blockIdx.x * blockDim.x + threadIdx.x;
  if (n >= Nn) return;
  const float* ap = agg + (long long)n * 64;
  float dn0 = denom[n * 2 + 0];
  float dn1 = denom[n * 2 + 1];
  float v[64];
  float sum = 0.f;
#pragma unroll
  for (int c = 0; c < 64; ++c) {
    float t = ap[c] / (c < 32 ? dn0 : dn1) + b[c];
    v[c] = t;
    sum += t;
  }
  float mu = sum * (1.f / 64.f);
  float sq = 0.f;
#pragma unroll
  for (int c = 0; c < 64; ++c) { float dd = v[c] - mu; sq += dd * dd; }
  float inv = rsqrtf(sq * (1.f / 64.f) + LN_EPS);
  float* xp = x2 + (long long)n * 64;
#pragma unroll
  for (int c = 0; c < 64; ++c) {
    xp[c] = elu((v[c] - mu) * inv * g[c] + be[c]);
  }
}

// ---------------------------------------------------------------------------
// layer-2 epilogue: normalize, +bias, LayerNorm(32), ELU, @Wout + bout -> out
// ---------------------------------------------------------------------------
__global__ void finalize2(const float* __restrict__ agg, const float* __restrict__ denom,
                          const float* __restrict__ b, const float* __restrict__ g,
                          const float* __restrict__ be, const float* __restrict__ Wout,
                          const float* __restrict__ bout, float* __restrict__ out, int Nn) {
  int n = blockIdx.x * blockDim.x + threadIdx.x;
  if (n >= Nn) return;
  const float* ap = agg + (long long)n * 32;
  float dn = denom[n];
  float v[32];
  float sum = 0.f;
#pragma unroll
  for (int c = 0; c < 32; ++c) {
    float t = ap[c] / dn + b[c];
    v[c] = t;
    sum += t;
  }
  float mu = sum * (1.f / 32.f);
  float sq = 0.f;
#pragma unroll
  for (int c = 0; c < 32; ++c) { float dd = v[c] - mu; sq += dd * dd; }
  float inv = rsqrtf(sq * (1.f / 32.f) + LN_EPS);
  float acc = 0.f;
#pragma unroll
  for (int c = 0; c < 32; ++c) {
    acc += elu((v[c] - mu) * inv * g[c] + be[c]) * Wout[c];
  }
  out[n] = acc + bout[0];
}

// ---------------------------------------------------------------------------
extern "C" void kernel_launch(void* const* d_in, const int* in_sizes, int n_in,
                              void* d_out, int out_size, void* d_ws, size_t ws_size,
                              hipStream_t stream) {
  const float*      x      = (const float*)d_in[0];
  const long long*  ei     = (const long long*)d_in[1];
  const float*      W1     = (const float*)d_in[2];
  const float*      a_src1 = (const float*)d_in[3];
  const float*      a_dst1 = (const float*)d_in[4];
  const float*      b1     = (const float*)d_in[5];
  const float*      g1     = (const float*)d_in[6];
  const float*      be1    = (const float*)d_in[7];
  const float*      W2     = (const float*)d_in[8];
  const float*      a_src2 = (const float*)d_in[9];
  const float*      a_dst2 = (const float*)d_in[10];
  const float*      b2     = (const float*)d_in[11];
  const float*      g2     = (const float*)d_in[12];
  const float*      be2    = (const float*)d_in[13];
  const float*      Wout   = (const float*)d_in[14];
  const float*      bout   = (const float*)d_in[15];
  float*            out    = (float*)d_out;

  const int Nn = in_sizes[0] / 64;   // 100000
  const int E  = in_sizes[1] / 2;    // 3200000

  // ---- workspace layout (all L2-friendly; edge stream is the only big HBM read)
  float* p = (float*)d_ws;
  float*    h1   = p; p += (size_t)Nn * 64;
  float*    x2   = p; p += (size_t)Nn * 64;
  float*    agg1 = p; p += (size_t)Nn * 64;
  float*    h2   = p; p += (size_t)Nn * 32;
  float*    agg2 = p; p += (size_t)Nn * 32;
  float*    als1 = p; p += (size_t)Nn * 2;
  float*    ald1 = p; p += (size_t)Nn * 2;
  float*    den1 = p; p += (size_t)Nn * 2;
  unsigned* mx1  = (unsigned*)p; p += (size_t)Nn * 2;
  float*    als2 = p; p += Nn;
  float*    ald2 = p; p += Nn;
  float*    den2 = p; p += Nn;
  unsigned* mx2  = (unsigned*)p; p += Nn;

  const int BLK = 256;  // 8 waves (wave32)

  // ===== layer 1 =====
  fill_u32<<<CDIV((long long)Nn * 2, BLK), BLK, 0, stream>>>(mx1, ENC_NEGINF, (long long)Nn * 2);
  fill_u32<<<CDIV((long long)Nn * 2, BLK), BLK, 0, stream>>>((unsigned*)den1, 0u, (long long)Nn * 2);
  fill_u32<<<CDIV((long long)Nn * 64, BLK), BLK, 0, stream>>>((unsigned*)agg1, 0u, (long long)Nn * 64);

  gemm_wmma<64><<<dim3(CDIV(Nn, 128), 4), BLK, 0, stream>>>(x, W1, h1, Nn);
  node_alpha<<<CDIV(Nn * 2, BLK), BLK, 0, stream>>>(h1, a_src1, a_dst1, als1, ald1, Nn, 2);

  long long tot1m = (long long)(E + Nn) * 2;
  edge_max<<<CDIV(tot1m, (long long)BLK), BLK, 0, stream>>>(ei, E, Nn, 2, als1, ald1, mx1);
  long long tot1a = (long long)(E + Nn) * 16;
  edge_acc<<<CDIV(tot1a, (long long)BLK), BLK, 0, stream>>>(ei, E, Nn, 2, als1, ald1, mx1, h1, agg1, den1);

  finalize1<<<CDIV(Nn, 128), 128, 0, stream>>>(agg1, den1, b1, g1, be1, x2, Nn);

  // ===== layer 2 =====
  fill_u32<<<CDIV((long long)Nn, BLK), BLK, 0, stream>>>(mx2, ENC_NEGINF, (long long)Nn);
  fill_u32<<<CDIV((long long)Nn, BLK), BLK, 0, stream>>>((unsigned*)den2, 0u, (long long)Nn);
  fill_u32<<<CDIV((long long)Nn * 32, BLK), BLK, 0, stream>>>((unsigned*)agg2, 0u, (long long)Nn * 32);

  gemm_wmma<32><<<dim3(CDIV(Nn, 128), 2), BLK, 0, stream>>>(x2, W2, h2, Nn);
  node_alpha<<<CDIV(Nn, BLK), BLK, 0, stream>>>(h2, a_src2, a_dst2, als2, ald2, Nn, 1);

  long long tot2m = (long long)(E + Nn);
  edge_max<<<CDIV(tot2m, (long long)BLK), BLK, 0, stream>>>(ei, E, Nn, 1, als2, ald2, mx2);
  long long tot2a = (long long)(E + Nn) * 8;
  edge_acc<<<CDIV(tot2a, (long long)BLK), BLK, 0, stream>>>(ei, E, Nn, 1, als2, ald2, mx2, h2, agg2, den2);

  finalize2<<<CDIV(Nn, 128), 128, 0, stream>>>(agg2, den2, b2, g2, be2, Wout, bout, out, Nn);
}
